// MultiHeadAttention_68066641707170
// MI455X (gfx1250) — matrix operand order repaired and checked by reading, not compiled, not run on hardware
//
#include <hip/hip_runtime.h>
#include <hip/hip_bf16.h>

// MHA forward for MI455X (gfx1250): all GEMMs via v_wmma_f32_16x16x32_f16
// (f32 accumulate). Full 16x2048 score strip lives in the 320KB CDNA5 LDS so
// the 537MB attention-weights tensor is written to HBM exactly once.
// Projection GEMMs are register-blocked 32x64 per wave (8 WMMA per 12 b128
// loads) to push the per-wave balance toward the WMMA issue rate.

typedef _Float16 half_t;
typedef __attribute__((ext_vector_type(16))) _Float16 v16h;
typedef __attribute__((ext_vector_type(8)))  _Float16 v8h;
typedef __attribute__((ext_vector_type(8)))  float    v8f;

#define D_MODEL 1024
#define S_LEN   2048
#define NTOK    4096   // 2 * 2048
#define D_HEAD  64

union V16H { v16h v; v8h h[2]; };

// A operand 16x32 f16, ISA layout: lane m<16 holds K {0..7,16..23},
// lane m+16 holds K {8..15,24..31}. base points at (row0, k0), row stride lda.
static __device__ inline v16h load_a_f16(const half_t* __restrict__ base, int lda, int lane) {
  int m = lane & 15, hi = lane >> 4;
  const half_t* p = base + m * lda + hi * 8;
  V16H r;
  r.h[0] = *(const v8h*)(p);
  r.h[1] = *(const v8h*)(p + 16);
  return r.v;
}

// B operand 32x16 f16 from Bt stored N-major (Bt[n][k] = B[k][n]):
// lane n<16 holds K {0..7,16..23}, lane n+16 holds K {8..15,24..31}. base at (k0, n0), stride ldb.
static __device__ inline v16h load_b_f16(const half_t* __restrict__ base, int ldb, int lane) {
  int n = lane & 15, hi = lane >> 4;
  const half_t* p = base + n * ldb + hi * 8;
  V16H r;
  r.h[0] = *(const v8h*)(p);
  r.h[1] = *(const v8h*)(p + 16);
  return r.v;
}

static __device__ inline v8f wmma_f16(v16h a, v16h b, v8f c) {
  return __builtin_amdgcn_wmma_f32_16x16x32_f16(false, a, false, b, (short)0, c, false, false);
}

// ---------------- prep ----------------
__global__ void cvt_x_kernel(const float* __restrict__ x, half_t* __restrict__ xh, int n) {
  int i = blockIdx.x * 256 + threadIdx.x;
  if (i < n) xh[i] = (half_t)x[i];
}

// wT[n*1024 + k] = (f16) w[k*1024 + n]   (one 1024x1024 matrix)
__global__ void cvt_wT_kernel(const float* __restrict__ w, half_t* __restrict__ wT) {
  int i = blockIdx.x * 256 + threadIdx.x;      // k fast -> coalesced writes
  int nn = i >> 10, kk = i & 1023;
  wT[i] = (half_t)w[kk * D_MODEL + nn];
}

// ---------------- QKV projection (register-blocked 32x64 per wave) ----------
// grid: (256, 3); 256 thr = 8 waves; each wave: 2 M-tiles x 4 N-tiles, K=1024.
__global__ void qkv_kernel(const half_t* __restrict__ xh,
                           const half_t* __restrict__ wT,   // [3][1024][1024] (q,k,v) transposed
                           const float* __restrict__ bq,
                           const float* __restrict__ bk,
                           const float* __restrict__ bv,
                           half_t* __restrict__ qbuf,       // [b][h][s][dh]
                           half_t* __restrict__ kbuf,       // [b][h][s][dh]
                           half_t* __restrict__ vtbuf) {    // [b][h][dh][s]
  int which = blockIdx.y;
  int lane = threadIdx.x & 31, wave = threadIdx.x >> 5;
  int tile = blockIdx.x * 8 + wave;            // 0..2047
  int mt = tile >> 4;                          // 0..127 -> 32-row macro tile
  int nt = tile & 15;                          // 0..15  -> 64-col macro tile
  int row0 = mt * 32, col0 = nt * 64;
  const half_t* a_base = xh + (size_t)row0 * D_MODEL;
  const half_t* b_base = wT + (size_t)which * D_MODEL * D_MODEL + (size_t)col0 * D_MODEL;

  v8f c00 = {}, c01 = {}, c02 = {}, c03 = {};
  v8f c10 = {}, c11 = {}, c12 = {}, c13 = {};
  for (int k0 = 0; k0 < D_MODEL; k0 += 32) {
    v16h a0 = load_a_f16(a_base + k0, D_MODEL, lane);
    v16h a1 = load_a_f16(a_base + (size_t)16 * D_MODEL + k0, D_MODEL, lane);
    v16h b0 = load_b_f16(b_base + (size_t)(0 * 16) * D_MODEL + k0, D_MODEL, lane);
    v16h b1 = load_b_f16(b_base + (size_t)(1 * 16) * D_MODEL + k0, D_MODEL, lane);
    v16h b2 = load_b_f16(b_base + (size_t)(2 * 16) * D_MODEL + k0, D_MODEL, lane);
    v16h b3 = load_b_f16(b_base + (size_t)(3 * 16) * D_MODEL + k0, D_MODEL, lane);
    c00 = wmma_f16(a0, b0, c00);  c01 = wmma_f16(a0, b1, c01);
    c02 = wmma_f16(a0, b2, c02);  c03 = wmma_f16(a0, b3, c03);
    c10 = wmma_f16(a1, b0, c10);  c11 = wmma_f16(a1, b1, c11);
    c12 = wmma_f16(a1, b2, c12);  c13 = wmma_f16(a1, b3, c13);
  }

  const float* bias = (which == 0) ? bq : ((which == 1) ? bk : bv);
  int n = lane & 15, hi = lane >> 4;
  v8f cr[2][4] = {{c00, c01, c02, c03}, {c10, c11, c12, c13}};
  for (int i = 0; i < 2; ++i) {
    for (int j = 0; j < 4; ++j) {
      int col = col0 + j * 16 + n;
      float bval = bias[col];
      int h = col >> 6, dh = col & 63;
      for (int r = 0; r < 8; ++r) {
        int t = row0 + i * 16 + r + 8 * hi;        // token 0..4095
        int bidx = t >> 11, s = t & 2047;
        float val = cr[i][j][r] + bval;
        size_t bh = (size_t)(bidx * 16 + h);
        if (which == 0)      qbuf[(bh * S_LEN + s) * D_HEAD + dh] = (half_t)val;
        else if (which == 1) kbuf[(bh * S_LEN + s) * D_HEAD + dh] = (half_t)val;
        else                 vtbuf[(bh * D_HEAD + dh) * S_LEN + s] = (half_t)val;
      }
    }
  }
}

// ---------------- attention core ----------------
// grid: (128 query tiles, 32 bh); 256 thr = 8 waves.
// LDS: scores f32[16][2048] | pbuf f16[16][2048] | acc f32[16][64] | red | stats
__global__ void attn_kernel(const half_t* __restrict__ qbuf,
                            const half_t* __restrict__ kbuf,
                            const half_t* __restrict__ vtbuf,
                            float* __restrict__ w_out,       // [b][h][s][s]
                            half_t* __restrict__ attnh) {    // [b*s][1024]
  extern __shared__ char smem_raw[];
  float*  scores = (float*)smem_raw;                              // 131072 B
  half_t* pbuf   = (half_t*)(smem_raw + 16 * S_LEN * 4);          //  65536 B
  float*  acc    = (float*)(smem_raw + 16 * S_LEN * 4 + 16 * S_LEN * 2); // 4096 B
  float*  red    = acc + 16 * 64;                                 //  1024 B
  float*  rowmax = red + 16 * 16;                                 //    64 B
  float*  rowsum = rowmax + 16;                                   //    64 B

  int qt = blockIdx.x, bh = blockIdx.y;
  int lane = threadIdx.x & 31, wave = threadIdx.x >> 5, tid = threadIdx.x;
  int q0 = qt * 16;
  const half_t* qbase = qbuf + ((size_t)bh * S_LEN + q0) * D_HEAD;
  const half_t* kbase = kbuf + (size_t)bh * S_LEN * D_HEAD;
  const half_t* vbase = vtbuf + (size_t)bh * D_HEAD * S_LEN;
  const float scale = 0.125f;                 // 1/sqrt(64)
  int n = lane & 15, hi = lane >> 4;

  // Q tile as two K=32 A-operands (dh = 64), reused for all key chunks.
  v16h aq0 = load_a_f16(qbase + 0,  D_HEAD, lane);
  v16h aq1 = load_a_f16(qbase + 32, D_HEAD, lane);

  // Phase 1: scores = scale * Q K^T  -> LDS strip [16][2048]
  for (int kt = 0; kt < 16; ++kt) {
    int kb = (wave * 16 + kt) * 16;           // key index base
    v16h b0 = load_b_f16(kbase + (size_t)kb * D_HEAD + 0,  D_HEAD, lane);
    v16h b1 = load_b_f16(kbase + (size_t)kb * D_HEAD + 32, D_HEAD, lane);
    v8f c = {};
    c = wmma_f16(aq0, b0, c);
    c = wmma_f16(aq1, b1, c);
    for (int r = 0; r < 8; ++r)
      scores[(r + 8 * hi) * S_LEN + kb + n] = c[r] * scale;
  }
  __syncthreads();

  // Phase 2: row softmax stats (16 rows x 16 partials)
  {
    int row = tid >> 4, sub = tid & 15;
    float m = -3.0e38f;
    for (int t = 0; t < 128; ++t) m = fmaxf(m, scores[row * S_LEN + sub + t * 16]);
    red[row * 16 + sub] = m;
    __syncthreads();
    if (sub == 0) {
      float mm = red[row * 16];
      for (int t = 1; t < 16; ++t) mm = fmaxf(mm, red[row * 16 + t]);
      rowmax[row] = mm;
    }
    __syncthreads();
    float mm = rowmax[row];
    float s = 0.f;
    for (int t = 0; t < 128; ++t) s += __expf(scores[row * S_LEN + sub + t * 16] - mm);
    red[row * 16 + sub] = s;
    __syncthreads();
    if (sub == 0) {
      float ss = 0.f;
      for (int t = 0; t < 16; ++t) ss += red[row * 16 + t];
      rowsum[row] = ss;
    }
    __syncthreads();
  }

  // Normalize: coalesced single write of weights to HBM + f16 copy for P.V
  {
    float* wrow = w_out + (size_t)bh * S_LEN * S_LEN + (size_t)q0 * S_LEN;
    for (int i = tid; i < 16 * S_LEN; i += 256) {
      int row = i >> 11, col = i & 2047;
      float p = __expf(scores[i] - rowmax[row]) / rowsum[row];
      pbuf[i] = (half_t)p;
      wrow[(size_t)row * S_LEN + col] = p;
    }
    for (int i = tid; i < 16 * 64; i += 256) acc[i] = 0.f;
  }
  __syncthreads();

  // Phase 3: attended = P * V; each wave owns 256 keys, reduce via ds_add_f32
  {
    v8f c0 = {}, c1 = {}, c2 = {}, c3 = {};
    for (int ks = 0; ks < 8; ++ks) {
      int kk = wave * 256 + ks * 32;
      v16h a  = load_a_f16(pbuf + kk, S_LEN, lane);
      v16h b0 = load_b_f16(vbase + (size_t)(0 * 16) * S_LEN + kk, S_LEN, lane);
      v16h b1 = load_b_f16(vbase + (size_t)(1 * 16) * S_LEN + kk, S_LEN, lane);
      v16h b2 = load_b_f16(vbase + (size_t)(2 * 16) * S_LEN + kk, S_LEN, lane);
      v16h b3 = load_b_f16(vbase + (size_t)(3 * 16) * S_LEN + kk, S_LEN, lane);
      c0 = wmma_f16(a, b0, c0);
      c1 = wmma_f16(a, b1, c1);
      c2 = wmma_f16(a, b2, c2);
      c3 = wmma_f16(a, b3, c3);
    }
    for (int r = 0; r < 8; ++r) {
      int rr = r + 8 * hi;
      atomicAdd(&acc[rr * 64 +  0 + n], c0[r]);
      atomicAdd(&acc[rr * 64 + 16 + n], c1[r]);
      atomicAdd(&acc[rr * 64 + 32 + n], c2[r]);
      atomicAdd(&acc[rr * 64 + 48 + n], c3[r]);
    }
  }
  __syncthreads();

  // Phase 4: attended tile -> f16 [4096][1024] (heads recombined)
  {
    int b = bh >> 4, h = bh & 15;
    for (int i = tid; i < 16 * 64; i += 256) {
      int r = i >> 6, col = i & 63;
      attnh[((size_t)(b * S_LEN + q0 + r)) * D_MODEL + h * 64 + col] = (half_t)acc[i];
    }
  }
}

// ---------------- output projection (register-blocked 32x64 per wave) -------
__global__ void out_proj_kernel(const half_t* __restrict__ attnh,
                                const half_t* __restrict__ woT,
                                const float* __restrict__ bo,
                                float* __restrict__ out) {
  int lane = threadIdx.x & 31, wave = threadIdx.x >> 5;
  int tile = blockIdx.x * 8 + wave;            // 0..2047
  int mt = tile >> 4, nt = tile & 15;
  int row0 = mt * 32, col0 = nt * 64;
  const half_t* a_base = attnh + (size_t)row0 * D_MODEL;
  const half_t* b_base = woT + (size_t)col0 * D_MODEL;

  v8f c00 = {}, c01 = {}, c02 = {}, c03 = {};
  v8f c10 = {}, c11 = {}, c12 = {}, c13 = {};
  for (int k0 = 0; k0 < D_MODEL; k0 += 32) {
    v16h a0 = load_a_f16(a_base + k0, D_MODEL, lane);
    v16h a1 = load_a_f16(a_base + (size_t)16 * D_MODEL + k0, D_MODEL, lane);
    v16h b0 = load_b_f16(b_base + (size_t)(0 * 16) * D_MODEL + k0, D_MODEL, lane);
    v16h b1 = load_b_f16(b_base + (size_t)(1 * 16) * D_MODEL + k0, D_MODEL, lane);
    v16h b2 = load_b_f16(b_base + (size_t)(2 * 16) * D_MODEL + k0, D_MODEL, lane);
    v16h b3 = load_b_f16(b_base + (size_t)(3 * 16) * D_MODEL + k0, D_MODEL, lane);
    c00 = wmma_f16(a0, b0, c00);  c01 = wmma_f16(a0, b1, c01);
    c02 = wmma_f16(a0, b2, c02);  c03 = wmma_f16(a0, b3, c03);
    c10 = wmma_f16(a1, b0, c10);  c11 = wmma_f16(a1, b1, c11);
    c12 = wmma_f16(a1, b2, c12);  c13 = wmma_f16(a1, b3, c13);
  }

  int n = lane & 15, hi = lane >> 4;
  v8f cr[2][4] = {{c00, c01, c02, c03}, {c10, c11, c12, c13}};
  for (int i = 0; i < 2; ++i) {
    for (int j = 0; j < 4; ++j) {
      int col = col0 + j * 16 + n;
      float bb = bo[col];
      for (int r = 0; r < 8; ++r)
        out[(size_t)(row0 + i * 16 + r + 8 * hi) * D_MODEL + col] = cr[i][j][r] + bb;
    }
  }
}

extern "C" void kernel_launch(void* const* d_in, const int* in_sizes, int n_in,
                              void* d_out, int out_size, void* d_ws, size_t ws_size,
                              hipStream_t stream) {
  const float* x  = (const float*)d_in[0];
  const float* wq = (const float*)d_in[1];
  const float* bq = (const float*)d_in[2];
  const float* wk = (const float*)d_in[3];
  const float* bk = (const float*)d_in[4];
  const float* wv = (const float*)d_in[5];
  const float* bv = (const float*)d_in[6];
  const float* wo = (const float*)d_in[7];
  const float* bo = (const float*)d_in[8];

  char* ws = (char*)d_ws;
  half_t* xh = (half_t*)ws;  ws += (size_t)NTOK * D_MODEL * 2;              // 8 MB
  half_t* wT = (half_t*)ws;  ws += (size_t)4 * D_MODEL * D_MODEL * 2;       // 8 MB (q,k,v,o)
  half_t* qb = (half_t*)ws;  ws += (size_t)NTOK * D_MODEL * 2;              // 8 MB
  half_t* kb = (half_t*)ws;  ws += (size_t)NTOK * D_MODEL * 2;              // 8 MB
  half_t* vt = (half_t*)ws;  ws += (size_t)NTOK * D_MODEL * 2;              // 8 MB
  half_t* ah = (half_t*)ws;  ws += (size_t)NTOK * D_MODEL * 2;              // 8 MB

  float* out_o = (float*)d_out;                          // [2,2048,1024]
  float* out_w = (float*)d_out + (size_t)NTOK * D_MODEL; // [2,16,2048,2048]

  // prep: f16 conversions + weight transposes
  cvt_x_kernel<<<(NTOK * D_MODEL) / 256, 256, 0, stream>>>(x, xh, NTOK * D_MODEL);
  const int WBLK = (D_MODEL * D_MODEL) / 256;
  cvt_wT_kernel<<<WBLK, 256, 0, stream>>>(wq, wT + 0ull * D_MODEL * D_MODEL);
  cvt_wT_kernel<<<WBLK, 256, 0, stream>>>(wk, wT + 1ull * D_MODEL * D_MODEL);
  cvt_wT_kernel<<<WBLK, 256, 0, stream>>>(wv, wT + 2ull * D_MODEL * D_MODEL);
  cvt_wT_kernel<<<WBLK, 256, 0, stream>>>(wo, wT + 3ull * D_MODEL * D_MODEL);

  // QKV: 128 x 16 macro tiles (32x64), 8 wave-tiles per block, 3 matrices
  qkv_kernel<<<dim3(256, 3), 256, 0, stream>>>(xh, wT, bq, bk, bv, qb, kb, vt);

  // attention: 128 query tiles x 32 (b,h); ~197 KB dynamic LDS (<320 KB WGP)
  size_t smem = (size_t)16 * S_LEN * 4 + 16 * S_LEN * 2 + 16 * 64 * 4 + 16 * 16 * 4 + 2 * 16 * 4;
  attn_kernel<<<dim3(128, 32), 256, smem, stream>>>(qb, kb, vt, out_w, ah);

  // out projection
  out_proj_kernel<<<256, 256, 0, stream>>>(ah, wT + 3ull * D_MODEL * D_MODEL, bo, out_o);
}
